// FlashRWLargeAttention_21861383536741
// MI455X (gfx1250) — compile-verified, matrix operand order checked by reading, not run on hardware
//
#include <hip/hip_runtime.h>
#include <hip/hip_bf16.h>

// ---------------- problem constants (match reference) ----------------
#define HIDDEN   4096
#define NGROUPS  4
#define NHG      8          // q heads per group
#define HEAD     128
#define QKV_N    5120       // NGROUPS * HEAD * (NHG + 2)
#define BATCH    4
#define SEQ      1024
#define TTOT     (BATCH * SEQ)   // 4096

typedef __bf16 v16bf  __attribute__((ext_vector_type(16)));
typedef __bf16 bf16x8 __attribute__((ext_vector_type(8)));
typedef float  v8f    __attribute__((ext_vector_type(8)));

static __device__ __forceinline__ v16bf cat8(bf16x8 lo, bf16x8 hi) {
    return __builtin_shufflevector(lo, hi, 0,1,2,3,4,5,6,7,8,9,10,11,12,13,14,15);
}

static __device__ __forceinline__ v8f wmma_bf16(v16bf a, v16bf b, v8f c) {
    return __builtin_amdgcn_wmma_f32_16x16x32_bf16(false, a, false, b, (short)0, c, false, false);
}

// ---------------- fp32 -> bf16 conversion ----------------
__global__ void f32_to_bf16_kernel(const float* __restrict__ in, __bf16* __restrict__ out, int n) {
    int i = blockIdx.x * blockDim.x + threadIdx.x;
    if (i < n) out[i] = (__bf16)in[i];
}

// ---------------- tiled bf16 WMMA GEMM: C(f32) = A(bf16,MxK) * B(bf16,KxN) ----------------
// BM=128 BN=128 BK=32, 256 threads = 8 waves, wave grid 2(M) x 4(N), each wave 64x32 = 4x2 tiles.
#define BM 128
#define BN 128
#define BK 32

__global__ __launch_bounds__(256) void gemm_bf16_f32(const __bf16* __restrict__ A,
                                                     const __bf16* __restrict__ B,
                                                     float* __restrict__ C,
                                                     int M, int N, int K) {
    __shared__ __align__(16) __bf16 lds_a[BM][BK];   // row-major (m,k)
    __shared__ __align__(16) __bf16 lds_b[BN][BK];   // transposed  (n,k)

    const int tid   = threadIdx.x;
    const int lane  = tid & 31;
    const int wave  = tid >> 5;
    const int wm    = wave >> 2;        // 0..1 : 64 rows
    const int wn    = wave & 3;         // 0..3 : 32 cols
    const int l16   = lane & 15;
    const int lhalf = lane >> 4;
    const int bm0   = blockIdx.y * BM;
    const int bn0   = blockIdx.x * BN;

    // per-thread staging coordinates (fixed across K loop)
    const int a_row = tid >> 1;
    const int a_col = (tid & 1) * 16;
    const int b_kk  = tid >> 3;
    const int b_nn  = (tid & 7) * 16;
    const __bf16* a_src = A + (size_t)(bm0 + a_row) * K + a_col;
    const __bf16* b_src = B + (size_t)b_kk * N + bn0 + b_nn;

    v8f acc[4][2] = {};

    for (int k0 = 0; k0 < K; k0 += BK) {
        // ---- stage A tile (each thread: one row-half, 16 elems via 2x b128) ----
        {
            const __bf16* src = a_src + k0;
            *(bf16x8*)&lds_a[a_row][a_col]     = *(const bf16x8*)src;
            *(bf16x8*)&lds_a[a_row][a_col + 8] = *(const bf16x8*)(src + 8);
        }
        // ---- stage B tile transposed: thread owns (k = tid>>3, 16 n's) ----
        {
            const __bf16* src = b_src + (size_t)k0 * N;
            bf16x8 b0 = *(const bf16x8*)src;
            bf16x8 b1 = *(const bf16x8*)(src + 8);
#pragma unroll
            for (int j = 0; j < 8; ++j) {
                lds_b[b_nn + j][b_kk]     = b0[j];
                lds_b[b_nn + 8 + j][b_kk] = b1[j];
            }
        }
        // ---- speculative prefetch of next K tile into L2 (global_prefetch_b8) ----
        if (k0 + BK < K) {
            __builtin_prefetch(a_src + (k0 + BK), 0, 1);
            __builtin_prefetch(b_src + (size_t)(k0 + BK) * N, 0, 1);
        }
        __syncthreads();

        // ---- fragments (A-layout: lane%16=row, contiguous K runs of 8 at lhalf*8 and 16+lhalf*8) ----
        v16bf afrag[4];
#pragma unroll
        for (int i = 0; i < 4; ++i) {
            const int row = wm * 64 + i * 16 + l16;
            bf16x8 lo = *(const bf16x8*)&lds_a[row][lhalf * 8];
            bf16x8 hi = *(const bf16x8*)&lds_a[row][16 + lhalf * 8];
            afrag[i] = cat8(lo, hi);
        }
        v16bf bfrag[2];
#pragma unroll
        for (int j = 0; j < 2; ++j) {
            const int col = wn * 32 + j * 16 + l16;
            bf16x8 lo = *(const bf16x8*)&lds_b[col][lhalf * 8];
            bf16x8 hi = *(const bf16x8*)&lds_b[col][16 + lhalf * 8];
            bfrag[j] = cat8(lo, hi);
        }
#pragma unroll
        for (int i = 0; i < 4; ++i)
#pragma unroll
            for (int j = 0; j < 2; ++j)
                acc[i][j] = wmma_bf16(afrag[i], bfrag[j], acc[i][j]);
        __syncthreads();
    }

    // ---- write back (C layout: VGPR r -> row r + 8*lhalf, col = lane%16) ----
#pragma unroll
    for (int i = 0; i < 4; ++i) {
#pragma unroll
        for (int j = 0; j < 2; ++j) {
            const int gm = bm0 + wm * 64 + i * 16 + 8 * lhalf;
            const int gn = bn0 + wn * 32 + j * 16 + l16;
            float* rowp = C + (size_t)gm * N + gn;
#pragma unroll
            for (int r = 0; r < 8; ++r) {
                *rowp = acc[i][j][r];
                rowp += N;
            }
        }
    }
}

// ---------------- RoPE + head split: qkv f32 [T][G][10][128] -> bf16 Q/K/V ----------------
// Q: [B][G][NHG][S][128], K/V: [B][G][S][128]
__global__ void rope_split_kernel(const float* __restrict__ qkv,
                                  const float* __restrict__ cosb,
                                  const float* __restrict__ sinb,
                                  __bf16* __restrict__ Qb,
                                  __bf16* __restrict__ Kb,
                                  __bf16* __restrict__ Vb) {
    const int idx = blockIdx.x * blockDim.x + threadIdx.x;  // TTOT*G*10*64
    if (idx >= TTOT * NGROUPS * 10 * 64) return;
    const int d  = idx & 63;
    const int hh = (idx >> 6) % 10;
    const int g  = (idx / 640) & (NGROUPS - 1);
    const int t  = idx / (NGROUPS * 10 * 64);
    const int b  = t / SEQ;
    const int s_ = t % SEQ;

    const float c  = cosb[t * 64 + d];
    const float sn = sinb[t * 64 + d];
    const float* src = qkv + ((size_t)(t * NGROUPS + g) * 10 + hh) * HEAD;
    const float x1 = src[d];
    const float x2 = src[64 + d];

    if (hh < NHG) {
        __bf16* dst = Qb + ((size_t)(((b * NGROUPS + g) * NHG + hh) * SEQ + s_)) * HEAD;
        dst[d]      = (__bf16)(x1 * c - x2 * sn);
        dst[64 + d] = (__bf16)(x2 * c + x1 * sn);
    } else if (hh == NHG) {
        __bf16* dst = Kb + ((size_t)((b * NGROUPS + g) * SEQ + s_)) * HEAD;
        dst[d]      = (__bf16)(x1 * c - x2 * sn);
        dst[64 + d] = (__bf16)(x2 * c + x1 * sn);
    } else {
        __bf16* dst = Vb + ((size_t)((b * NGROUPS + g) * SEQ + s_)) * HEAD;
        dst[d]      = (__bf16)x1;
        dst[64 + d] = (__bf16)x2;
    }
}

// ---------------- flash attention (causal, per-head), bf16 WMMA ----------------
// Block: 256 threads = 8 waves, one head, 128 q-rows (16 per wave). Key blocks of 32.
__global__ __launch_bounds__(256) void flash_attn_kernel(const __bf16* __restrict__ Qb,
                                                         const __bf16* __restrict__ Kb,
                                                         const __bf16* __restrict__ Vb,
                                                         __bf16* __restrict__ Out) {
    __shared__ __align__(16) __bf16 lds_k[32][HEAD];      // (key, d)      8KB
    __shared__ __align__(16) __bf16 lds_vt[HEAD][32];     // (d, key)      8KB
    __shared__ __align__(16) __bf16 lds_p[8][16][32];     // per-wave P    8KB

    const int tid   = threadIdx.x;
    const int lane  = tid & 31;
    const int wave  = tid >> 5;
    const int l16   = lane & 15;
    const int lhalf = lane >> 4;

    const int bx   = blockIdx.x;
    const int qblk = bx & 7;           // 8 q-blocks of 128 rows
    const int head = bx >> 3;          // 0..127
    const int b    = head >> 5;
    const int g    = (head >> 3) & 3;
    const int h    = head & 7;
    const int q0   = qblk * 128 + wave * 16;

    const __bf16* Qh = Qb + (size_t)(((b * NGROUPS + g) * NHG + h) * SEQ) * HEAD;
    const __bf16* Kh = Kb + (size_t)((b * NGROUPS + g) * SEQ) * HEAD;
    const __bf16* Vh = Vb + (size_t)((b * NGROUPS + g) * SEQ) * HEAD;

    // Q fragments: 16 rows x 128 (4 chunks of K=32)
    v16bf qf[4];
    {
        const __bf16* qrow = Qh + (size_t)(q0 + l16) * HEAD;
#pragma unroll
        for (int c = 0; c < 4; ++c) {
            bf16x8 lo = *(const bf16x8*)(qrow + c * 32 + lhalf * 8);
            bf16x8 hi = *(const bf16x8*)(qrow + c * 32 + 16 + lhalf * 8);
            qf[c] = cat8(lo, hi);
        }
    }

    v8f o[8] = {};
    float m[8], l[8];
#pragma unroll
    for (int r = 0; r < 8; ++r) { m[r] = -1e30f; l[r] = 0.0f; }

    const float scale = 0.08838834764831845f;   // 1/sqrt(128)
    const int kb_end = (qblk * 128 + 127) >> 5; // inclusive causal limit for the block

    // fixed staging coordinates
    const int kk = tid >> 3;            // 0..31
    const int dd = (tid & 7) * 16;      // 0..112

    for (int kb = 0; kb <= kb_end; ++kb) {
        __syncthreads();
        // ---- cooperative stage of K (row-major) and V (transposed) ----
        {
            const __bf16* ks = Kh + (size_t)(kb * 32 + kk) * HEAD + dd;
            *(bf16x8*)&lds_k[kk][dd]     = *(const bf16x8*)ks;
            *(bf16x8*)&lds_k[kk][dd + 8] = *(const bf16x8*)(ks + 8);
            const __bf16* vs = Vh + (size_t)(kb * 32 + kk) * HEAD + dd;
            bf16x8 v0 = *(const bf16x8*)vs;
            bf16x8 v1 = *(const bf16x8*)(vs + 8);
#pragma unroll
            for (int j = 0; j < 8; ++j) {
                lds_vt[dd + j][kk]     = v0[j];
                lds_vt[dd + 8 + j][kk] = v1[j];
            }
        }
        // ---- speculative prefetch of next key block into L2 ----
        if (kb < kb_end) {
            __builtin_prefetch(Kh + (size_t)((kb + 1) * 32 + kk) * HEAD + dd, 0, 1);
            __builtin_prefetch(Vh + (size_t)((kb + 1) * 32 + kk) * HEAD + dd, 0, 1);
        }
        __syncthreads();
        if (kb * 32 > q0 + 15) continue;        // fully masked for this wave

        // ---- scores: S = Q (16x128) * K^T (128x32), two 16x16 accum tiles ----
        v8f s0 = {}, s1 = {};
#pragma unroll
        for (int c = 0; c < 4; ++c) {
            bf16x8 a0 = *(const bf16x8*)&lds_k[l16][c * 32 + lhalf * 8];
            bf16x8 a1 = *(const bf16x8*)&lds_k[l16][c * 32 + 16 + lhalf * 8];
            v16bf kf0 = cat8(a0, a1);
            bf16x8 b0 = *(const bf16x8*)&lds_k[16 + l16][c * 32 + lhalf * 8];
            bf16x8 b1 = *(const bf16x8*)&lds_k[16 + l16][c * 32 + 16 + lhalf * 8];
            v16bf kf1 = cat8(b0, b1);
            s0 = wmma_bf16(qf[c], kf0, s0);
            s1 = wmma_bf16(qf[c], kf1, s1);
        }

        // ---- online softmax (per-lane rows r + 8*lhalf, cols l16 / l16+16) ----
        const int col0 = kb * 32 + l16;
#pragma unroll
        for (int r = 0; r < 8; ++r) {
            const int row = q0 + r + 8 * lhalf;
            float v0s = s0[r] * scale; if (col0 > row)       v0s = -1e30f;
            float v1s = s1[r] * scale; if (col0 + 16 > row)  v1s = -1e30f;
            float mx = fmaxf(v0s, v1s);
#pragma unroll
            for (int off = 1; off < 16; off <<= 1) mx = fmaxf(mx, __shfl_xor(mx, off, 32));
            const float mnew = fmaxf(m[r], mx);
            const float corr = __expf(m[r] - mnew);
            const float p0 = __expf(v0s - mnew);
            const float p1 = __expf(v1s - mnew);
            float rs = p0 + p1;
#pragma unroll
            for (int off = 1; off < 16; off <<= 1) rs += __shfl_xor(rs, off, 32);
            l[r] = l[r] * corr + rs;
            m[r] = mnew;
#pragma unroll
            for (int t = 0; t < 8; ++t) o[t][r] = o[t][r] * corr;
            lds_p[wave][r + 8 * lhalf][l16]      = (__bf16)p0;
            lds_p[wave][r + 8 * lhalf][16 + l16] = (__bf16)p1;
        }
        asm volatile("s_wait_dscnt 0" ::: "memory");   // P stores visible before A-layout reload

        // ---- reload P as 16x32 bf16 A-fragment, then O += P * V ----
        bf16x8 plo = *(const bf16x8*)&lds_p[wave][l16][lhalf * 8];
        bf16x8 phi = *(const bf16x8*)&lds_p[wave][l16][16 + lhalf * 8];
        v16bf pf = cat8(plo, phi);
#pragma unroll
        for (int t = 0; t < 8; ++t) {
            bf16x8 vlo = *(const bf16x8*)&lds_vt[t * 16 + l16][lhalf * 8];
            bf16x8 vhi = *(const bf16x8*)&lds_vt[t * 16 + l16][16 + lhalf * 8];
            v16bf vf = cat8(vlo, vhi);
            o[t] = wmma_bf16(pf, vf, o[t]);
        }
    }

    // ---- epilogue: normalize, write bf16 attn output [T][4096] (g,h,d order) ----
#pragma unroll
    for (int r = 0; r < 8; ++r) {
        const float inv = 1.0f / l[r];
        const int trow = b * SEQ + q0 + r + 8 * lhalf;
        __bf16* op = Out + (size_t)trow * HIDDEN + (size_t)((g * NHG + h) * HEAD) + l16;
#pragma unroll
        for (int t = 0; t < 8; ++t)
            op[t * 16] = (__bf16)(o[t][r] * inv);
    }
}

// ---------------- host orchestration ----------------
extern "C" void kernel_launch(void* const* d_in, const int* in_sizes, int n_in,
                              void* d_out, int out_size, void* d_ws, size_t ws_size,
                              hipStream_t stream) {
    const float* hidden  = (const float*)d_in[0];
    const float* cosb    = (const float*)d_in[1];
    const float* sinb    = (const float*)d_in[2];
    // d_in[3] cu_seqlens, d_in[4] max_s, d_in[5] layer_past: unused (fixed shapes)
    const float* w_qkv   = (const float*)d_in[6];
    const float* w_dense = (const float*)d_in[7];
    float* out = (float*)d_out;

    char* ws = (char*)d_ws;
    size_t off = 0;
    auto alloc = [&](size_t bytes) { size_t p = off; off = (off + bytes + 255) & ~(size_t)255; return p; };

    const size_t q_elems  = (size_t)BATCH * NGROUPS * NHG * SEQ * HEAD;  // 16,777,216
    const size_t kv_elems = (size_t)BATCH * NGROUPS * SEQ * HEAD;        //  2,097,152

    __bf16* hs_b   = (__bf16*)(ws + alloc((size_t)TTOT * HIDDEN * 2));
    __bf16* wqkv_b = (__bf16*)(ws + alloc((size_t)HIDDEN * QKV_N * 2));
    __bf16* wd_b   = (__bf16*)(ws + alloc((size_t)HIDDEN * HIDDEN * 2));
    float*  qkv_f  = (float*)(ws + alloc((size_t)TTOT * QKV_N * 4));
    __bf16* Qb     = (__bf16*)(ws + alloc(q_elems * 2));
    __bf16* Kb     = (__bf16*)(ws + alloc(kv_elems * 2));
    __bf16* Vb     = (__bf16*)(ws + alloc(kv_elems * 2));
    __bf16* attn_b = (__bf16*)(ws + alloc((size_t)TTOT * HIDDEN * 2));
    (void)ws_size; (void)in_sizes; (void)n_in; (void)out_size;

    // 1) convert activations + weights to bf16
    {
        int n = TTOT * HIDDEN;
        f32_to_bf16_kernel<<<(n + 255) / 256, 256, 0, stream>>>(hidden, hs_b, n);
        n = HIDDEN * QKV_N;
        f32_to_bf16_kernel<<<(n + 255) / 256, 256, 0, stream>>>(w_qkv, wqkv_b, n);
        n = HIDDEN * HIDDEN;
        f32_to_bf16_kernel<<<(n + 255) / 256, 256, 0, stream>>>(w_dense, wd_b, n);
    }
    // 2) QKV projection: (T x 4096) x (4096 x 5120)
    {
        dim3 grid(QKV_N / BN, TTOT / BM);
        gemm_bf16_f32<<<grid, 256, 0, stream>>>(hs_b, wqkv_b, qkv_f, TTOT, QKV_N, HIDDEN);
    }
    // 3) RoPE + split to head-major bf16 Q/K/V
    {
        int n = TTOT * NGROUPS * 10 * 64;
        rope_split_kernel<<<(n + 255) / 256, 256, 0, stream>>>(qkv_f, cosb, sinb, Qb, Kb, Vb);
    }
    // 4) causal flash attention, 128 heads x 8 q-blocks
    {
        dim3 grid(128 * 8);
        flash_attn_kernel<<<grid, 256, 0, stream>>>(Qb, Kb, Vb, attn_b);
    }
    // 5) output projection: (T x 4096) x (4096 x 4096) -> fp32 out
    {
        dim3 grid(HIDDEN / BN, TTOT / BM);
        gemm_bf16_f32<<<grid, 256, 0, stream>>>(attn_b, wd_b, out, TTOT, HIDDEN, HIDDEN);
    }
}
